// DualFeatureExtractionStructureBlock_64587718197526
// MI455X (gfx1250) — compile-verified, matrix operand order checked by reading, not run on hardware
//
#include <hip/hip_runtime.h>
#include <hip/hip_bf16.h>

typedef __attribute__((ext_vector_type(16))) _Float16 v16h;
typedef __attribute__((ext_vector_type(8)))  _Float16 v8h;
typedef __attribute__((ext_vector_type(8)))  float    v8f;

#define BB 32
#define SS 256
#define NV 10
#define DD 128
#define D3 384
#define BN (BB*NV)          // 320
#define TT (BN*SS)          // 81920 tokens (b*N,s)
#define TB (BB*SS)          // 8192 tokens (b,s)

__device__ __forceinline__ float gelu_exact(float x) {
    return 0.5f * x * (1.0f + erff(x * 0.70710678118654752440f));
}

// For 16-bit WMMA fragments (wave32, 16x16x32): lane half selects base8 = 0 or 8;
// halves 0..7 cover K = base8+0..7, halves 8..15 cover K = 16+base8+0..7.
// => fragment = two contiguous 16B runs when K is the contiguous axis.
__device__ __forceinline__ v16h load_frag(const _Float16* p) {
    v8h lo = *(const v8h*)p;
    v8h hi = *(const v8h*)(p + 16);
    return __builtin_shufflevector(lo, hi, 0,1,2,3,4,5,6,7,8,9,10,11,12,13,14,15);
}

// Raw LDS byte offset of a __shared__ object (addrspace(3) ptrtoint).
__device__ __forceinline__ unsigned lds_offset(const void* p) {
    return (unsigned)(unsigned long long)(__attribute__((address_space(3))) const void*)p;
}

// Async global->LDS 16B copy (GLOBAL_LOAD_ASYNC_TO_LDS_B128, ASYNCcnt-tracked).
__device__ __forceinline__ void async_copy_b128(unsigned lds_off, const void* g) {
    unsigned long long ga = (unsigned long long)g;
    asm volatile("global_load_async_to_lds_b128 %0, %1, off" :: "v"(lds_off), "v"(ga) : "memory");
}

// ---------------------------------------------------------------------------
// Weight prep: f32 [K,N] -> f16 transposed [N,K]  (K becomes contiguous)
// ---------------------------------------------------------------------------
__global__ void cvt_tr_f16(const float* __restrict__ src, _Float16* __restrict__ dst,
                           int K, int N) {
    int i = blockIdx.x * blockDim.x + threadIdx.x;
    if (i < K * N) {
        int n = i / K, kk = i % K;
        dst[i] = (_Float16)src[(size_t)kk * N + n];
    }
}

// ---------------------------------------------------------------------------
// Conv stems: 4 in-ch, 3 taps, 128 out-ch, pad=1, + exact GELU, store f16
// ---------------------------------------------------------------------------
__global__ void conv_tgt_kernel(const float* __restrict__ x, const float* __restrict__ w,
                                const float* __restrict__ b, _Float16* __restrict__ out) {
    int t = blockIdx.x;               // b*S + s
    int bb = t / SS, s = t % SS;
    int d = threadIdx.x;              // 0..127
    float acc = b[d];
    #pragma unroll
    for (int ci = 0; ci < 4; ++ci)
        #pragma unroll
        for (int tap = 0; tap < 3; ++tap) {
            int sp = s + tap - 1;
            if (sp >= 0 && sp < SS)
                acc += x[((size_t)bb * SS + sp) * 44 + ci] * w[d * 12 + ci * 3 + tap];
        }
    out[(size_t)t * DD + d] = (_Float16)gelu_exact(acc);
}

__global__ void conv_arr_kernel(const float* __restrict__ x, const float* __restrict__ w,
                                const float* __restrict__ b, _Float16* __restrict__ out) {
    int t = blockIdx.x;               // bn*S + s
    int bn = t / SS, s = t % SS;
    int bb = bn / NV, nv = bn % NV;
    int d = threadIdx.x;
    float acc = b[d];
    #pragma unroll
    for (int ci = 0; ci < 4; ++ci)
        #pragma unroll
        for (int tap = 0; tap < 3; ++tap) {
            int sp = s + tap - 1;
            if (sp >= 0 && sp < SS)
                acc += x[((size_t)bb * SS + sp) * 44 + 4 + nv * 4 + ci] * w[d * 12 + ci * 3 + tap];
        }
    out[(size_t)t * DD + d] = (_Float16)gelu_exact(acc);
}

// ---------------------------------------------------------------------------
// Generic WMMA GEMM: C[M,N] = epi(A[M,K] @ Wt^T + bias), Wt is [N,K] (K-contig)
// block = 128 threads (4 waves); block tile 64x64; wave tile 16x64.
// The 64x32 weight tile shared by all 4 waves is staged into LDS with
// double-buffered GLOBAL_LOAD_ASYNC_TO_LDS_B128 (ASYNCcnt), read as ds_load_b128.
// epi 0: v = (acc+bias)*alpha            -> Cf (f32) and/or Ch (f16)
// epi 1: v = gelu(acc+bias)*gate[m,n]    -> Ch (gated MLP)
// epi 2: v = acc+bias -> Ch stored transposed per 256-row sequence:
//        Ch[( (m/256)*N + n )*256 + m%256]   (for V of attention)
// ---------------------------------------------------------------------------
__global__ void gemm_wmma_kernel(const _Float16* __restrict__ A,
                                 const _Float16* __restrict__ Wt,
                                 const float* __restrict__ bias,
                                 const _Float16* __restrict__ gate,
                                 float* __restrict__ Cf,
                                 _Float16* __restrict__ Ch,
                                 int M, int K, int Nn, int epi, float alpha) {
    __shared__ _Float16 btile[2][64 * 32];   // [n][k] per buffer, 4KB each
    const int wave = threadIdx.x >> 5;
    const int lane = threadIdx.x & 31;
    const int half = lane >> 4;
    const int base8 = half * 8;
    const int rowBase = blockIdx.y * 64 + wave * 16;
    const int colBase = blockIdx.x * 64;

    v8f acc[4] = {v8f{}, v8f{}, v8f{}, v8f{}};
    const int m = rowBase + (lane & 15);
    const _Float16* arow = A + (size_t)m * K + base8;

    // Stage one 64x32 weight tile: 256 16B chunks, 2 per thread.
    auto stage = [&](int buf, int k0) {
        #pragma unroll
        for (int i = 0; i < 2; ++i) {
            int c = threadIdx.x + i * 128;         // chunk id 0..255
            int n = c >> 2;                        // 0..63
            int kq = (c & 3) * 8;                  // 0,8,16,24 (halves)
            unsigned loff = lds_offset(&btile[buf][n * 32 + kq]);
            async_copy_b128(loff, Wt + (size_t)(colBase + n) * K + k0 + kq);
        }
    };

    stage(0, 0);
    int buf = 0;
    for (int k0 = 0; k0 < K; k0 += 32) {
        const bool more = (k0 + 32 < K);
        if (more) {
            stage(buf ^ 1, k0 + 32);
            asm volatile("s_wait_asynccnt 0x2" ::: "memory");  // prior tile landed
        } else {
            asm volatile("s_wait_asynccnt 0x0" ::: "memory");  // last tile landed
        }
        __syncthreads();
        if (more)
            __builtin_prefetch(arow + k0 + 32, 0, 1);
        v16h af = load_frag(arow + k0);
        #pragma unroll
        for (int t = 0; t < 4; ++t) {
            v16h bf = load_frag(&btile[buf][(t * 16 + (lane & 15)) * 32 + base8]);
            acc[t] = __builtin_amdgcn_wmma_f32_16x16x32_f16(
                false, af, false, bf, (short)0, acc[t], false, false);
        }
        __syncthreads();   // protect buf before it is re-staged next iter
        buf ^= 1;
    }

    #pragma unroll
    for (int t = 0; t < 4; ++t) {
        const int n = colBase + t * 16 + (lane & 15);
        const float bv = bias[n];
        #pragma unroll
        for (int r = 0; r < 8; ++r) {
            const int mr = rowBase + r + 8 * half;
            float v = acc[t][r] + bv;
            if (epi == 1) v = gelu_exact(v) * (float)gate[(size_t)mr * Nn + n];
            v *= alpha;
            if (epi == 2) {
                Ch[((size_t)(mr >> 8) * Nn + n) * SS + (mr & 255)] = (_Float16)v;
            } else {
                if (Cf) Cf[(size_t)mr * Nn + n] = v;
                if (Ch) Ch[(size_t)mr * Nn + n] = (_Float16)v;
            }
        }
    }
}

// ---------------------------------------------------------------------------
// concat c = [xm, ym, xm*ym] (f16)
// ---------------------------------------------------------------------------
__global__ void build_c_kernel(const _Float16* __restrict__ xmh, const _Float16* __restrict__ ymh,
                               _Float16* __restrict__ ch) {
    int t = blockIdx.x;               // bn*S + s
    int bn = t / SS, s = t % SS, bb = bn / NV;
    int j = threadIdx.x;
    float xm = (float)xmh[((size_t)bb * SS + s) * DD + j];
    float ym = (float)ymh[(size_t)t * DD + j];
    size_t o = (size_t)t * D3;
    ch[o + j]       = (_Float16)xm;
    ch[o + 128 + j] = (_Float16)ym;
    ch[o + 256 + j] = (_Float16)(xm * ym);
}

// ---------------------------------------------------------------------------
// LN1: h = LN(gelu(u) + c), eps 1e-6; c recomputed from xm/ym. Writes f32+f16.
// ---------------------------------------------------------------------------
__global__ void ln1_kernel(const float* __restrict__ u, const _Float16* __restrict__ xmh,
                           const _Float16* __restrict__ ymh, const float* __restrict__ g,
                           const float* __restrict__ be, float* __restrict__ hf,
                           _Float16* __restrict__ hh) {
    __shared__ float s1[128], s2[128];
    int t = blockIdx.x;
    int bn = t / SS, s = t % SS, bb = bn / NV;
    int j = threadIdx.x;
    float xm = (float)xmh[((size_t)bb * SS + s) * DD + j];
    float ym = (float)ymh[(size_t)t * DD + j];
    const float* ur = u + (size_t)t * D3;
    float r0 = gelu_exact(ur[j])       + xm;
    float r1 = gelu_exact(ur[128 + j]) + ym;
    float r2 = gelu_exact(ur[256 + j]) + xm * ym;
    s1[j] = r0 + r1 + r2;
    s2[j] = r0 * r0 + r1 * r1 + r2 * r2;
    __syncthreads();
    for (int o = 64; o > 0; o >>= 1) {
        if (j < o) { s1[j] += s1[j + o]; s2[j] += s2[j + o]; }
        __syncthreads();
    }
    float mean = s1[0] * (1.0f / D3);
    float var  = s2[0] * (1.0f / D3) - mean * mean;
    float inv  = rsqrtf(var + 1e-6f);
    size_t o0 = (size_t)t * D3;
    float v0 = (r0 - mean) * inv * g[j]       + be[j];
    float v1 = (r1 - mean) * inv * g[128 + j] + be[128 + j];
    float v2 = (r2 - mean) * inv * g[256 + j] + be[256 + j];
    hf[o0 + j] = v0;       hh[o0 + j]       = (_Float16)v0;
    hf[o0 + 128 + j] = v1; hh[o0 + 128 + j] = (_Float16)v1;
    hf[o0 + 256 + j] = v2; hh[o0 + 256 + j] = (_Float16)v2;
}

// ---------------------------------------------------------------------------
// LN2: h2 = LN(h + u), eps 1e-5. Writes f16 only.
// ---------------------------------------------------------------------------
__global__ void ln2_kernel(const float* __restrict__ hf, const float* __restrict__ u,
                           const float* __restrict__ g, const float* __restrict__ be,
                           _Float16* __restrict__ h2h) {
    __shared__ float s1[128], s2[128];
    int t = blockIdx.x;
    int j = threadIdx.x;
    size_t o0 = (size_t)t * D3;
    float r0 = hf[o0 + j]       + u[o0 + j];
    float r1 = hf[o0 + 128 + j] + u[o0 + 128 + j];
    float r2 = hf[o0 + 256 + j] + u[o0 + 256 + j];
    s1[j] = r0 + r1 + r2;
    s2[j] = r0 * r0 + r1 * r1 + r2 * r2;
    __syncthreads();
    for (int o = 64; o > 0; o >>= 1) {
        if (j < o) { s1[j] += s1[j + o]; s2[j] += s2[j + o]; }
        __syncthreads();
    }
    float mean = s1[0] * (1.0f / D3);
    float var  = s2[0] * (1.0f / D3) - mean * mean;
    float inv  = rsqrtf(var + 1e-5f);
    h2h[o0 + j]       = (_Float16)((r0 - mean) * inv * g[j]       + be[j]);
    h2h[o0 + 128 + j] = (_Float16)((r1 - mean) * inv * g[128 + j] + be[128 + j]);
    h2h[o0 + 256 + j] = (_Float16)((r2 - mean) * inv * g[256 + j] + be[256 + j]);
}

// ---------------------------------------------------------------------------
// Flash-style attention: one block per (bn, head); 8 waves; wave owns 16-row
// query tiles; 32-key steps; scores (8 wmma) + P@V (8 wmma) per step; online
// softmax. All fragments are 2x16B contiguous loads: q/k are d-contiguous,
// v is pre-transposed [bn, chan, key]. P staged as f16 in per-wave LDS and
// re-read as ds_load_b128 A-fragments. q is pre-scaled by 1/sqrt(dk).
// ---------------------------------------------------------------------------
__global__ void attn_kernel(const _Float16* __restrict__ q, const _Float16* __restrict__ k,
                            const _Float16* __restrict__ vt, _Float16* __restrict__ ao) {
    __shared__ _Float16 pbuf[8][16 * 32];
    const int bh = blockIdx.x;
    const int bn = bh / 3, head = bh % 3;
    const size_t base  = (size_t)bn * SS * D3 + (size_t)head * DD;       // q/k/ao rows
    const size_t vbase = ((size_t)bn * D3 + (size_t)head * DD) * SS;     // vt channels
    const int wave = threadIdx.x >> 5;
    const int lane = threadIdx.x & 31;
    const int half = lane >> 4;
    const int base8 = half * 8;
    _Float16* P = pbuf[wave];

    for (int qt = wave; qt < SS / 16; qt += 8) {
        const int q0 = qt * 16;
        const _Float16* qrow = q + base + (size_t)(q0 + (lane & 15)) * D3 + base8;
        v16h qf[4];
        #pragma unroll
        for (int kc = 0; kc < 4; ++kc)
            qf[kc] = load_frag(qrow + kc * 32);

        v8f o[8] = {v8f{}, v8f{}, v8f{}, v8f{}, v8f{}, v8f{}, v8f{}, v8f{}};
        float rmax[8], rsum[8];
        #pragma unroll
        for (int r = 0; r < 8; ++r) { rmax[r] = -1e30f; rsum[r] = 0.0f; }

        for (int s0 = 0; s0 < SS; s0 += 32) {
            v8f sc0 = v8f{}, sc1 = v8f{};
            const _Float16* krow0 = k + base + (size_t)(s0 + (lane & 15)) * D3 + base8;
            const _Float16* krow1 = krow0 + (size_t)16 * D3;
            #pragma unroll
            for (int kc = 0; kc < 4; ++kc) {
                v16h kf0 = load_frag(krow0 + kc * 32);
                v16h kf1 = load_frag(krow1 + kc * 32);
                sc0 = __builtin_amdgcn_wmma_f32_16x16x32_f16(false, qf[kc], false, kf0, (short)0, sc0, false, false);
                sc1 = __builtin_amdgcn_wmma_f32_16x16x32_f16(false, qf[kc], false, kf1, (short)0, sc1, false, false);
            }
            // online softmax update (rows live in lane halves; reduce across 16 lanes)
            float mx[8];
            #pragma unroll
            for (int r = 0; r < 8; ++r) mx[r] = fmaxf(sc0[r], sc1[r]);
            #pragma unroll
            for (int off = 1; off < 16; off <<= 1)
                #pragma unroll
                for (int r = 0; r < 8; ++r)
                    mx[r] = fmaxf(mx[r], __shfl_xor(mx[r], off, 32));
            float scl[8];
            #pragma unroll
            for (int r = 0; r < 8; ++r) {
                float nm = fmaxf(rmax[r], mx[r]);
                scl[r] = __expf(rmax[r] - nm);
                rmax[r] = nm;
            }
            #pragma unroll
            for (int r = 0; r < 8; ++r) {
                sc0[r] = __expf(sc0[r] - rmax[r]);
                sc1[r] = __expf(sc1[r] - rmax[r]);
            }
            float sm[8];
            #pragma unroll
            for (int r = 0; r < 8; ++r) sm[r] = sc0[r] + sc1[r];
            #pragma unroll
            for (int off = 1; off < 16; off <<= 1)
                #pragma unroll
                for (int r = 0; r < 8; ++r)
                    sm[r] += __shfl_xor(sm[r], off, 32);
            #pragma unroll
            for (int r = 0; r < 8; ++r) rsum[r] = rsum[r] * scl[r] + sm[r];
            #pragma unroll
            for (int t = 0; t < 8; ++t)
                #pragma unroll
                for (int r = 0; r < 8; ++r)
                    o[t][r] *= scl[r];

            // stage P (C layout, f16) -> re-read as A fragment (16 q-rows x 32 keys)
            #pragma unroll
            for (int r = 0; r < 8; ++r) {
                int rr = r + 8 * half;
                P[rr * 32 + (lane & 15)]      = (_Float16)sc0[r];
                P[rr * 32 + 16 + (lane & 15)] = (_Float16)sc1[r];
            }
            __asm__ volatile("s_wait_dscnt 0" ::: "memory");
            v16h pf = load_frag(P + (lane & 15) * 32 + base8);
            __asm__ volatile("s_wait_dscnt 0" ::: "memory");

            #pragma unroll
            for (int t = 0; t < 8; ++t) {
                const int col = t * 16 + (lane & 15);
                v16h vf = load_frag(vt + vbase + (size_t)col * SS + s0 + base8);
                o[t] = __builtin_amdgcn_wmma_f32_16x16x32_f16(false, pf, false, vf, (short)0, o[t], false, false);
            }
        }
        #pragma unroll
        for (int t = 0; t < 8; ++t) {
            const int col = t * 16 + (lane & 15);
            #pragma unroll
            for (int r = 0; r < 8; ++r) {
                int mr = q0 + r + 8 * half;
                ao[base + (size_t)mr * D3 + col] = (_Float16)(o[t][r] / rsum[r]);
            }
        }
    }
}

// ---------------------------------------------------------------------------
// Final: out[b,s,o] = sum_n sum_c fused[b,n,s,c] * conv4_w[o,c,0,n] + conv4_b[o]
// ---------------------------------------------------------------------------
__global__ void out_kernel(const float* __restrict__ fsd, const float* __restrict__ w4,
                           const float* __restrict__ b4, float* __restrict__ out) {
    __shared__ float sh[NV * DD];
    int t = blockIdx.x;               // b*S + s
    int bb = t / SS, s = t % SS;
    int o = threadIdx.x;
    for (int i = threadIdx.x; i < NV * DD; i += 128) {
        int nv = i >> 7, c = i & 127;
        sh[i] = fsd[(((size_t)(bb * NV + nv)) * SS + s) * DD + c];
    }
    __syncthreads();
    float acc = b4[o];
    #pragma unroll 2
    for (int nv = 0; nv < NV; ++nv)
        for (int c = 0; c < DD; ++c)
            acc += sh[nv * DD + c] * w4[((size_t)o * DD + c) * NV + nv];
    out[(size_t)t * DD + o] = acc;
}

// ---------------------------------------------------------------------------
extern "C" void kernel_launch(void* const* d_in, const int* in_sizes, int n_in,
                              void* d_out, int out_size, void* d_ws, size_t ws_size,
                              hipStream_t stream) {
    const float* x       = (const float*)d_in[0];
    const float* conv1_w = (const float*)d_in[1];
    const float* conv1_b = (const float*)d_in[2];
    const float* conv2_w = (const float*)d_in[3];
    const float* conv2_b = (const float*)d_in[4];
    const float* conv4_w = (const float*)d_in[5];
    const float* conv4_b = (const float*)d_in[6];
    const float* fc1_w   = (const float*)d_in[7];
    const float* fc1_b   = (const float*)d_in[8];
    const float* fc2_w   = (const float*)d_in[9];
    const float* fc2_b   = (const float*)d_in[10];
    const float* fus_w   = (const float*)d_in[11];
    const float* fus_b   = (const float*)d_in[12];
    const float* ln1_g   = (const float*)d_in[13];
    const float* ln1_b   = (const float*)d_in[14];
    const float* wq      = (const float*)d_in[15];
    const float* bq      = (const float*)d_in[16];
    const float* wk      = (const float*)d_in[17];
    const float* bk      = (const float*)d_in[18];
    const float* wv      = (const float*)d_in[19];
    const float* bv      = (const float*)d_in[20];
    const float* wo      = (const float*)d_in[21];
    const float* bo      = (const float*)d_in[22];
    const float* ln2_g   = (const float*)d_in[23];
    const float* ln2_b   = (const float*)d_in[24];
    const float* fco_w   = (const float*)d_in[25];
    const float* fco_b   = (const float*)d_in[26];
    float* out = (float*)d_out;

    // ---- workspace layout (bump allocator, 256B aligned) ----
    size_t off = 0;
    char* ws = (char*)d_ws;
    auto take = [&](size_t bytes) -> char* {
        char* p = ws + off;
        off += (bytes + 255) & ~(size_t)255;
        return p;
    };
    _Float16* x1h = (_Float16*)take((size_t)TB * DD * 2);
    _Float16* x2h = (_Float16*)take((size_t)TT * DD * 2);
    _Float16* xmh = (_Float16*)take((size_t)TB * DD * 2);
    _Float16* ymh = (_Float16*)take((size_t)TT * DD * 2);
    _Float16* ch  = (_Float16*)take((size_t)TT * D3 * 2);   // aliased as ao later
    float*    uf  = (float*)   take((size_t)TT * D3 * 4);   // GEMM f32 scratch
    float*    hf  = (float*)   take((size_t)TT * D3 * 4);
    _Float16* hh  = (_Float16*)take((size_t)TT * D3 * 2);   // aliased as h2 later
    _Float16* qh  = (_Float16*)take((size_t)TT * D3 * 2);   // aliased as fused (f32) later
    _Float16* kh  = (_Float16*)take((size_t)TT * D3 * 2);
    _Float16* vth = (_Float16*)take((size_t)TT * D3 * 2);   // V stored transposed
    _Float16* fc1t = (_Float16*)take((size_t)DD * DD * 2);  // all weights: [N,K] f16
    _Float16* fc2t = (_Float16*)take((size_t)DD * DD * 2);
    _Float16* fust = (_Float16*)take((size_t)D3 * D3 * 2);
    _Float16* wqt  = (_Float16*)take((size_t)D3 * D3 * 2);
    _Float16* wkt  = (_Float16*)take((size_t)D3 * D3 * 2);
    _Float16* wvt  = (_Float16*)take((size_t)D3 * D3 * 2);
    _Float16* wot  = (_Float16*)take((size_t)D3 * D3 * 2);
    _Float16* fcot = (_Float16*)take((size_t)D3 * DD * 2);
    _Float16* aoh  = ch;              // c dead after fus GEMM
    _Float16* h2h  = hh;              // h_f16 dead after q/k/v GEMMs
    float*    fsd  = (float*)qh;      // q dead after attention

    auto cvt = [&](const float* s, _Float16* d, int K, int N) {
        cvt_tr_f16<<<(K * N + 255) / 256, 256, 0, stream>>>(s, d, K, N);
    };
    cvt(fc1_w, fc1t, DD, DD);
    cvt(fc2_w, fc2t, DD, DD);
    cvt(fus_w, fust, D3, D3);
    cvt(wq,   wqt,  D3, D3);
    cvt(wk,   wkt,  D3, D3);
    cvt(wv,   wvt,  D3, D3);
    cvt(wo,   wot,  D3, D3);
    cvt(fco_w, fcot, D3, DD);

    auto gemm = [&](const _Float16* A, const _Float16* Wt, const float* bias,
                    const _Float16* gate, float* Cf, _Float16* Ch,
                    int M, int K, int Nn, int epi, float alpha) {
        dim3 grid(Nn / 64, M / 64);
        gemm_wmma_kernel<<<grid, 128, 0, stream>>>(A, Wt, bias, gate, Cf, Ch, M, K, Nn, epi, alpha);
    };

    // conv stems + GELU
    conv_tgt_kernel<<<TB, DD, 0, stream>>>(x, conv1_w, conv1_b, x1h);
    conv_arr_kernel<<<TT, DD, 0, stream>>>(x, conv2_w, conv2_b, x2h);

    // gated MLPs: xm = gelu(x1@fc1+b)*x1 ; ym = gelu(x2@fc2+b)*x2
    gemm(x1h, fc1t, fc1_b, x1h, nullptr, xmh, TB, DD, DD, 1, 1.0f);
    gemm(x2h, fc2t, fc2_b, x2h, nullptr, ymh, TT, DD, DD, 1, 1.0f);

    // c = concat(xm, ym, xm*ym); fus GEMM; LN1
    build_c_kernel<<<TT, DD, 0, stream>>>(xmh, ymh, ch);
    gemm(ch, fust, fus_b, nullptr, uf, nullptr, TT, D3, D3, 0, 1.0f);
    ln1_kernel<<<TT, DD, 0, stream>>>(uf, xmh, ymh, ln1_g, ln1_b, hf, hh);

    // q/k/v projections (1/sqrt(128) folded into q; V stored transposed)
    const float qscale = 0.08838834764831845f;
    gemm(hh, wqt, bq, nullptr, nullptr, qh, TT, D3, D3, 0, qscale);
    gemm(hh, wkt, bk, nullptr, nullptr, kh, TT, D3, D3, 0, 1.0f);
    gemm(hh, wvt, bv, nullptr, nullptr, vth, TT, D3, D3, 2, 1.0f);

    // flash attention: 320 sequences x 3 heads
    attn_kernel<<<BN * 3, 256, 0, stream>>>(qh, kh, vth, aoh);

    // output projection + LN2 + final 384->128 projection
    gemm(aoh, wot, bo, nullptr, uf, nullptr, TT, D3, D3, 0, 1.0f);
    ln2_kernel<<<TT, DD, 0, stream>>>(hf, uf, ln2_g, ln2_b, h2h);
    gemm(h2h, fcot, fco_b, nullptr, fsd, nullptr, TT, D3, DD, 0, 1.0f);

    // aggregate neighbors with conv4 weights
    out_kernel<<<TB, DD, 0, stream>>>(fsd, conv4_w, conv4_b, out);
}
